// multiheadlatentattention_13761075216516
// MI455X (gfx1250) — matrix operand order repaired and checked by reading, not compiled, not run on hardware
//
#include <hip/hip_runtime.h>

// ---------- feature gates (device pass only; safe fallbacks otherwise) ------

#if defined(__HIP_DEVICE_COMPILE__)
#  if __has_builtin(__builtin_amdgcn_tensor_load_to_lds) && \
      __has_builtin(__builtin_amdgcn_s_wait_tensorcnt)
#    define STAGE_TDM_OK 1
#    if (__clang_major__ >= 23)
#      define TDM_6ARG 1            // upstream clang-23: 6-arg builtin form
#    endif
#  endif
#  if __has_builtin(__builtin_amdgcn_global_load_async_to_lds_b128) && \
      __has_builtin(__builtin_amdgcn_s_wait_asynccnt)
#    define STAGE_ASYNC_OK 1
#  endif
#endif

#define GAS __attribute__((address_space(1)))
#define LAS __attribute__((address_space(3)))

// ---------- types / helpers -------------------------------------------------

typedef __attribute__((ext_vector_type(16))) __bf16 v16bf;
typedef __attribute__((ext_vector_type(8)))  float  v8f;
typedef __attribute__((ext_vector_type(4)))  unsigned int u32x4;
typedef __attribute__((ext_vector_type(8)))  int i32x8;
typedef __attribute__((ext_vector_type(4)))  int i32x4;
typedef __attribute__((ext_vector_type(4)))  int v4i;

struct alignas(16) U4 { unsigned int x, y, z, w; };
struct alignas(32) U8 { U4 lo, hi; };

__device__ __forceinline__ v16bf mkfrag(U4 lo, U4 hi) {
  U8 t{lo, hi};
  return __builtin_bit_cast(v16bf, t);
}

__device__ __forceinline__ unsigned short f2bf(float f) {
  unsigned int u = __builtin_bit_cast(unsigned int, f);
  u += 0x7FFFu + ((u >> 16) & 1u);          // round-to-nearest-even
  return (unsigned short)(u >> 16);
}

__device__ __forceinline__ v8f v8zero() {
  v8f z = {0.f, 0.f, 0.f, 0.f, 0.f, 0.f, 0.f, 0.f};
  return z;
}

// generic LDS pointer -> raw 32-bit LDS byte offset
__device__ __forceinline__ unsigned lds_addr32(const void* p) {
  return (unsigned)(unsigned long long)(LAS const char*)p;
}

// one 16B global->LDS tile element (async if available)
__device__ __forceinline__ void ld_g2l_128(void* l, const void* g) {
#if defined(STAGE_ASYNC_OK)
  __builtin_amdgcn_global_load_async_to_lds_b128((GAS v4i*)g, (LAS v4i*)l, 0, 0);
#else
  *(U4*)l = *(const U4*)g;
#endif
}

#if defined(STAGE_TDM_OK)
// 2-D bf16 tile DMA via Tensor Data Mover: tile_w elems/row, tile_h rows,
// row stride in elems; pad_iv/pad_am per D# encoding (DWORD LDS padding).
__device__ __forceinline__ void tdm_load_2d(void* lds, const void* g,
                                            unsigned tile_w, unsigned tile_h,
                                            unsigned long long row_stride,
                                            unsigned pad_iv, unsigned pad_am) {
  unsigned long long ga = (unsigned long long)(size_t)g;
  unsigned la = lds_addr32(lds);
  u32x4 g0 = { 1u,                                   // count=1, user mode
               la,                                   // lds_addr
               (unsigned)ga,                         // global_addr[31:0]
               (unsigned)(ga >> 32) | (2u << 30) };  // global_addr[56:32] | type=2
  unsigned w0 = (1u << 16)                           // data_size = 2 bytes
              | (1u << 20)                           // pad_enable
              | (pad_iv << 22) | (pad_am << 25);
  i32x8 g1 = { (int)w0,
               (int)((tile_w & 0xFFFFu) << 16),                       // tensor_dim0 lo
               (int)((tile_w >> 16) | ((tile_h & 0xFFFFu) << 16)),    // dim0 hi | dim1 lo
               (int)((tile_h >> 16) | ((tile_w & 0xFFFFu) << 16)),    // dim1 hi | tile_dim0
               (int)(tile_h & 0xFFFFu),                               // tile_dim1 | tile_dim2=0
               (int)(unsigned)(row_stride & 0xFFFFFFFFull),           // dim0_stride lo
               (int)(unsigned)(row_stride >> 32),                     // dim0_stride hi
               0 };
  i32x4 gz = { 0, 0, 0, 0 };
#if defined(TDM_6ARG)
  i32x8 gz8 = { 0, 0, 0, 0, 0, 0, 0, 0 };
  __builtin_amdgcn_tensor_load_to_lds(g0, g1, gz, gz, gz8, 0);
#else
  __builtin_amdgcn_tensor_load_to_lds(g0, g1, gz, gz, 0);
#endif
}
#endif

// ---------- f32 -> bf16 conversion (plain) ----------------------------------

__global__ void cvt_f32_bf16(const float* __restrict__ in,
                             unsigned short* __restrict__ out, int n) {
  int i = blockIdx.x * blockDim.x + threadIdx.x;
  if (i < n) out[i] = f2bf(in[i]);
}

// ---------- f32 [K][N] -> bf16 transposed [N][K] (batched over z) -----------

__global__ void __launch_bounds__(256)
cvt_f32_bf16_T(const float* __restrict__ in, unsigned short* __restrict__ out,
               int K, int N) {
  in  += (size_t)blockIdx.z * (size_t)K * (size_t)N;
  out += (size_t)blockIdx.z * (size_t)K * (size_t)N;
  __shared__ float tile[32][33];
  const int k0 = blockIdx.y * 32, n0 = blockIdx.x * 32;
  const int tx = threadIdx.x & 31, ty = threadIdx.x >> 5;  // 32 x 8
#pragma unroll
  for (int r = 0; r < 32; r += 8)
    tile[ty + r][tx] = in[(size_t)(k0 + ty + r) * (size_t)N + n0 + tx];
  __syncthreads();
#pragma unroll
  for (int r = 0; r < 32; r += 8)
    out[(size_t)(n0 + ty + r) * (size_t)K + k0 + tx] = f2bf(tile[tx][ty + r]);
}

// ---------- bf16 WMMA GEMM:  C[M,N] = A[M,K] @ W[K,N] + bias ---------------
// W supplied PRE-TRANSPOSED as WT[N][K] so both tiles stage contiguously.
// block: 256 threads (8 wave32), tile 128x128, K step 32, double-buffered LDS,
// staged by TDM / async-to-LDS / plain b128.
// c_mode: 0 = f32 row-major, 1 = bf16 row-major, 2 = bf16 transposed [N][M].

__global__ void __launch_bounds__(256)
gemm_bf16_wmma(const unsigned short* __restrict__ A,
               const unsigned short* __restrict__ WT,
               const float* __restrict__ bias,
               void* __restrict__ Cv,
               int M, int N, int K,
               long long sA, long long sW, long long sBias, long long sC,
               int c_mode) {
  const int bz = blockIdx.z;
  A    += (size_t)bz * (size_t)sA;
  WT   += (size_t)bz * (size_t)sW;
  bias += (size_t)bz * (size_t)sBias;

  const int m0 = blockIdx.y * 128;
  const int n0 = blockIdx.x * 128;

  __shared__ alignas(16) unsigned short lA[2][128 * 40];  // [m][k] pitch 40
  __shared__ alignas(16) unsigned short lB[2][128 * 40];  // [n][k] pitch 40

  const int tid  = threadIdx.x;
  const int lane = tid & 31;
  const int wid  = tid >> 5;
  const int hl   = lane >> 4;
  const int l16  = lane & 15;
  const int wm   = wid >> 1;      // 0..3
  const int wn   = wid & 1;       // 0..1

  v8f acc[2][4];
#pragma unroll
  for (int i = 0; i < 2; ++i)
#pragma unroll
    for (int j = 0; j < 4; ++j) acc[i][j] = v8zero();

  auto loadTiles = [&](int kt, int buf) {
    const int k0 = kt * 32;
#if defined(STAGE_TDM_OK)
    // row = 32 elems = 16 DW; pad 4 DW -> pitch 20 DW = 40 halfwords
    if (wid == 0)
      tdm_load_2d(&lA[buf][0], A + (size_t)m0 * (size_t)K + k0, 32, 128,
                  (unsigned long long)K, 3u /*16 DW*/, 3u /*4 DW*/);
    else if (wid == 1)
      tdm_load_2d(&lB[buf][0], WT + (size_t)n0 * (size_t)K + k0, 32, 128,
                  (unsigned long long)K, 3u, 3u);
#else
    const int row  = tid >> 1;
    const int colh = (tid & 1) * 16;
    const unsigned short* ga = A  + (size_t)(m0 + row) * (size_t)K + k0 + colh;
    ld_g2l_128(&lA[buf][row * 40 + colh],     ga);
    ld_g2l_128(&lA[buf][row * 40 + colh + 8], ga + 8);
    const unsigned short* gb = WT + (size_t)(n0 + row) * (size_t)K + k0 + colh;
    ld_g2l_128(&lB[buf][row * 40 + colh],     gb);
    ld_g2l_128(&lB[buf][row * 40 + colh + 8], gb + 8);
#endif
  };
  auto stageWait = [&]() {
#if defined(STAGE_TDM_OK)
    if (wid <= 1) __builtin_amdgcn_s_wait_tensorcnt(0);
#elif defined(STAGE_ASYNC_OK)
    __builtin_amdgcn_s_wait_asynccnt(0);
#endif
  };

  const int nk = K >> 5;
  loadTiles(0, 0);
  stageWait();
  __syncthreads();

  for (int kt = 0; kt < nk; ++kt) {
    const int buf = kt & 1;
    if (kt + 1 < nk) loadTiles(kt + 1, buf ^ 1);

    v16bf af[2], bfm[4];
#pragma unroll
    for (int mr = 0; mr < 2; ++mr) {
      int row = wm * 32 + mr * 16 + l16;
      const unsigned short* p = &lA[buf][row * 40 + hl * 8];
      af[mr] = mkfrag(*(const U4*)p, *(const U4*)(p + 16));
    }
#pragma unroll
    for (int nc = 0; nc < 4; ++nc) {
      int col = wn * 64 + nc * 16 + l16;
      const unsigned short* p = &lB[buf][col * 40 + hl * 8];
      bfm[nc] = mkfrag(*(const U4*)p, *(const U4*)(p + 16));
    }
#pragma unroll
    for (int mr = 0; mr < 2; ++mr)
#pragma unroll
      for (int nc = 0; nc < 4; ++nc)
        acc[mr][nc] = __builtin_amdgcn_wmma_f32_16x16x32_bf16(
            false, af[mr], false, bfm[nc], (short)0, acc[mr][nc], false, false);

    stageWait();
    __syncthreads();
  }

  // Epilogue. C layout: lanes<16 rows j, lanes>=16 rows 8+j; col = l16.
  if (c_mode == 2) {                       // bf16, transposed [N][M], b128 packed
    unsigned short* C = (unsigned short*)Cv + (size_t)bz * (size_t)sC;
#pragma unroll
    for (int mr = 0; mr < 2; ++mr)
#pragma unroll
      for (int nc = 0; nc < 4; ++nc) {
        int colg = n0 + wn * 64 + nc * 16 + l16;
        float bv = bias[colg];
        unsigned int hw[4];
#pragma unroll
        for (int p = 0; p < 4; ++p) {
          unsigned short a = f2bf(acc[mr][nc][2 * p]     + bv);
          unsigned short b = f2bf(acc[mr][nc][2 * p + 1] + bv);
          hw[p] = (unsigned)a | ((unsigned)b << 16);
        }
        U4 pack{hw[0], hw[1], hw[2], hw[3]};
        int rowg0 = m0 + wm * 32 + mr * 16 + hl * 8;
        *(U4*)&C[(size_t)colg * (size_t)M + rowg0] = pack;
      }
  } else if (c_mode == 1) {                // bf16 row-major
    unsigned short* C = (unsigned short*)Cv + (size_t)bz * (size_t)sC;
#pragma unroll
    for (int mr = 0; mr < 2; ++mr)
#pragma unroll
      for (int nc = 0; nc < 4; ++nc) {
        int colg = n0 + wn * 64 + nc * 16 + l16;
        float bv = bias[colg];
#pragma unroll
        for (int j = 0; j < 8; ++j) {
          int rowg = m0 + wm * 32 + mr * 16 + hl * 8 + j;
          C[(size_t)rowg * (size_t)N + colg] = f2bf(acc[mr][nc][j] + bv);
        }
      }
  } else {                                 // f32 row-major
    float* C = (float*)Cv + (size_t)bz * (size_t)sC;
#pragma unroll
    for (int mr = 0; mr < 2; ++mr)
#pragma unroll
      for (int nc = 0; nc < 4; ++nc) {
        int colg = n0 + wn * 64 + nc * 16 + l16;
        float bv = bias[colg];
#pragma unroll
        for (int j = 0; j < 8; ++j) {
          int rowg = m0 + wm * 32 + mr * 16 + hl * 8 + j;
          C[(size_t)rowg * (size_t)N + colg] = acc[mr][nc][j] + bv;
        }
      }
  }
}

// ---------- causal flash attention (bf16 WMMA, fp32 online softmax) ---------
// grid: (S/64, H, B); block: 128 threads (4 wave32).
// V supplied pre-transposed per head: VT[h][dh][B*S] -> contiguous V-tile DMA.

__global__ void __launch_bounds__(128)
mla_flash_wmma(const unsigned short* __restrict__ Q,    // [B*S][2048]
               const unsigned short* __restrict__ Kh,   // [H][B*S][128]
               const unsigned short* __restrict__ VT,   // [H][128][B*S]
               unsigned short* __restrict__ O,          // [B*S][2048]
               int S, int BS) {
  const int h  = blockIdx.y;
  const int b  = blockIdx.z;
  const int q0 = blockIdx.x * 64;

  const int tid  = threadIdx.x;
  const int lane = tid & 31;
  const int w    = tid >> 5;
  const int hl   = lane >> 4;
  const int l16  = lane & 15;
  const int qw   = q0 + w * 16;

  __shared__ alignas(16) unsigned short vt[128 * 72];     // [dh][key], pitch 72
  __shared__ alignas(16) unsigned short pb[4 * 16 * 72];  // [wave][row][key]

  // Q fragments (A layout), straight from global
  v16bf qf[4];
  {
    const unsigned short* qp = Q + ((size_t)(b * S + qw + l16)) * 2048 + h * 128;
#pragma unroll
    for (int kc = 0; kc < 4; ++kc) {
      const unsigned short* p = qp + kc * 32 + hl * 8;
      qf[kc] = mkfrag(*(const U4*)p, *(const U4*)(p + 16));
    }
  }
  const unsigned short* Kp  = Kh + ((size_t)h * BS + (size_t)b * S) * 128;
  const unsigned short* VpT = VT + ((size_t)h * BS) * 128 + (size_t)b * S;

  float m_[8], l_[8];
  v8f oacc[8];
#pragma unroll
  for (int j = 0; j < 8; ++j) { m_[j] = -3.0e38f; l_[j] = 0.f; }
#pragma unroll
  for (int nt = 0; nt < 8; ++nt) oacc[nt] = v8zero();

  const int nblk = q0 / 64 + 1;
  for (int jb = 0; jb < nblk; ++jb) {
    const int j0 = jb * 64;

    __syncthreads();              // prior P@V done reading vt
#if defined(STAGE_TDM_OK)
    // row = 64 elems = 32 DW; pad 4 DW -> pitch 36 DW = 72 halfwords
    if (w == 0) {
      tdm_load_2d(vt, VpT + j0, 64, 128, (unsigned long long)BS,
                  4u /*32 DW*/, 3u /*4 DW*/);
      __builtin_amdgcn_s_wait_tensorcnt(0);
    }
#else
    {
      const int dh = tid;         // 128 rows, one per thread
      const unsigned short* src = VpT + (size_t)dh * (size_t)BS + j0;
#pragma unroll
      for (int c = 0; c < 8; ++c)
        ld_g2l_128(&vt[dh * 72 + c * 8], src + c * 8);
#if defined(STAGE_ASYNC_OK)
      __builtin_amdgcn_s_wait_asynccnt(0);
#endif
    }
#endif
    __syncthreads();

    // scores = Q @ K^T for 4 key tiles of 16 (K frags straight from global)
    v8f sacc[4];
#pragma unroll
    for (int kt = 0; kt < 4; ++kt) sacc[kt] = v8zero();
#pragma unroll
    for (int kt = 0; kt < 4; ++kt) {
      const unsigned short* kp = Kp + (size_t)(j0 + kt * 16 + l16) * 128;
#pragma unroll
      for (int kc = 0; kc < 4; ++kc) {
        const unsigned short* p = kp + kc * 32 + hl * 8;
        v16bf kf = mkfrag(*(const U4*)p, *(const U4*)(p + 16));
        sacc[kt] = __builtin_amdgcn_wmma_f32_16x16x32_bf16(
            false, qf[kc], false, kf, (short)0, sacc[kt], false, false);
      }
    }

    const float scale = 0.08838834764831845f;  // 1/sqrt(128)
#pragma unroll
    for (int kt = 0; kt < 4; ++kt)
#pragma unroll
      for (int j = 0; j < 8; ++j) {
        float s = sacc[kt][j] * scale;
        int keyg = j0 + kt * 16 + l16;
        int rowg = qw + hl * 8 + j;
        if (keyg > rowg) s = -3.0e38f;          // causal mask
        sacc[kt][j] = s;
      }

    // online softmax: half-wave (16-lane) row reductions
#pragma unroll
    for (int j = 0; j < 8; ++j) {
      float mx = sacc[0][j];
#pragma unroll
      for (int kt = 1; kt < 4; ++kt) mx = fmaxf(mx, sacc[kt][j]);
#pragma unroll
      for (int d = 1; d < 16; d <<= 1) mx = fmaxf(mx, __shfl_xor(mx, d, 32));
      float mn    = fmaxf(m_[j], mx);
      float alpha = __expf(m_[j] - mn);
      float rs    = 0.f;
#pragma unroll
      for (int kt = 0; kt < 4; ++kt) {
        float p = __expf(sacc[kt][j] - mn);
        rs += p;
        pb[(w * 16 + hl * 8 + j) * 72 + kt * 16 + l16] = f2bf(p);
      }
#pragma unroll
      for (int d = 1; d < 16; d <<= 1) rs += __shfl_xor(rs, d, 32);
      l_[j] = l_[j] * alpha + rs;
      m_[j] = mn;
#pragma unroll
      for (int nt = 0; nt < 8; ++nt) oacc[nt][j] = oacc[nt][j] * alpha;
    }

    // O += P @ V
#pragma unroll
    for (int kc = 0; kc < 2; ++kc) {
      const unsigned short* pp = &pb[(w * 16 + l16) * 72 + kc * 32 + hl * 8];
      v16bf pf = mkfrag(*(const U4*)pp, *(const U4*)(pp + 16));
#pragma unroll
      for (int nt = 0; nt < 8; ++nt) {
        const unsigned short* vp2 = &vt[(nt * 16 + l16) * 72 + kc * 32 + hl * 8];
        v16bf vf = mkfrag(*(const U4*)vp2, *(const U4*)(vp2 + 16));
        oacc[nt] = __builtin_amdgcn_wmma_f32_16x16x32_bf16(
            false, pf, false, vf, (short)0, oacc[nt], false, false);
      }
    }
  }

  // normalize and write O (bf16, heads packed back into [B*S][2048])
#pragma unroll
  for (int nt = 0; nt < 8; ++nt)
#pragma unroll
    for (int j = 0; j < 8; ++j) {
      float v = oacc[nt][j] / l_[j];
      int rowg = b * S + qw + hl * 8 + j;
      int colg = h * 128 + nt * 16 + l16;
      O[(size_t)rowg * 2048 + colg] = f2bf(v);
    }
}

// ---------- host orchestration ---------------------------------------------

extern "C" void kernel_launch(void* const* d_in, const int* in_sizes, int n_in,
                              void* d_out, int out_size, void* d_ws, size_t ws_size,
                              hipStream_t stream) {
  (void)in_sizes; (void)n_in; (void)out_size; (void)ws_size;
  const int B = 2, S = 2048, DIN = 2048, DOUT = 2048, DL = 512, H = 16, DH = 128;
  const int BS = B * S;  // 4096

  const float* x     = (const float*)d_in[0];
  const float* wq_w  = (const float*)d_in[1];
  const float* wq_b  = (const float*)d_in[2];
  const float* wk_w  = (const float*)d_in[3];
  const float* wk_b  = (const float*)d_in[4];
  const float* wkl_w = (const float*)d_in[5];
  const float* wkl_b = (const float*)d_in[6];
  const float* wv_w  = (const float*)d_in[7];
  const float* wv_b  = (const float*)d_in[8];
  const float* wvl_w = (const float*)d_in[9];
  const float* wvl_b = (const float*)d_in[10];
  const float* upk_w = (const float*)d_in[11];
  const float* upk_b = (const float*)d_in[12];
  const float* upv_w = (const float*)d_in[13];
  const float* upv_b = (const float*)d_in[14];
  const float* wo_w  = (const float*)d_in[15];
  const float* wo_b  = (const float*)d_in[16];

  char* ws = (char*)d_ws;
  size_t off = 0;
  auto alloc = [&](size_t elems) -> unsigned short* {
    unsigned short* p = (unsigned short*)(ws + off);
    off += ((elems * 2 + 255) & ~(size_t)255);
    return p;
  };
  unsigned short* xb    = alloc((size_t)BS * DIN);
  unsigned short* wqT   = alloc((size_t)DIN * DOUT);   // [DOUT][DIN]
  unsigned short* wkT   = alloc((size_t)DIN * DOUT);
  unsigned short* wvT   = alloc((size_t)DIN * DOUT);
  unsigned short* woT   = alloc((size_t)DOUT * DOUT);
  unsigned short* wklT  = alloc((size_t)DOUT * DL);    // [DL][DOUT]
  unsigned short* wvlT  = alloc((size_t)DOUT * DL);
  unsigned short* upkT  = alloc((size_t)H * DL * DH);  // [H][DH][DL]
  unsigned short* upvT  = alloc((size_t)H * DL * DH);
  unsigned short* qb    = alloc((size_t)BS * DOUT);
  unsigned short* ktmp  = alloc((size_t)BS * DOUT);
  unsigned short* vtmp  = alloc((size_t)BS * DOUT);
  unsigned short* klat  = alloc((size_t)BS * DL);
  unsigned short* vlat  = alloc((size_t)BS * DL);
  unsigned short* khead = alloc((size_t)H * BS * DH);  // [H][BS][DH]
  unsigned short* vheadT= alloc((size_t)H * BS * DH);  // [H][DH][BS]
  unsigned short* ob    = alloc((size_t)BS * DOUT);

  cvt_f32_bf16<<<dim3((unsigned)(((size_t)BS * DIN + 255) / 256)), dim3(256), 0,
                 stream>>>(x, xb, BS * DIN);

  auto cvtT = [&](const float* s, unsigned short* d, int K, int N, int batch) {
    cvt_f32_bf16_T<<<dim3(N / 32, K / 32, batch), dim3(256), 0, stream>>>(s, d, K, N);
  };
  cvtT(wq_w,  wqT,  DIN,  DOUT, 1);
  cvtT(wk_w,  wkT,  DIN,  DOUT, 1);
  cvtT(wv_w,  wvT,  DIN,  DOUT, 1);
  cvtT(wo_w,  woT,  DOUT, DOUT, 1);
  cvtT(wkl_w, wklT, DOUT, DL,   1);
  cvtT(wvl_w, wvlT, DOUT, DL,   1);
  cvtT(upk_w, upkT, DL,   DH,   H);
  cvtT(upv_w, upvT, DL,   DH,   H);

  const dim3 blk(256);
  auto grid = [](int N, int M, int z) { return dim3(N / 128, M / 128, z); };

  // q / k / v projections (bf16 row-major out)
  gemm_bf16_wmma<<<grid(DOUT, BS, 1), blk, 0, stream>>>(
      xb, wqT, wq_b, qb, BS, DOUT, DIN, 0, 0, 0, 0, 1);
  gemm_bf16_wmma<<<grid(DOUT, BS, 1), blk, 0, stream>>>(
      xb, wkT, wk_b, ktmp, BS, DOUT, DIN, 0, 0, 0, 0, 1);
  gemm_bf16_wmma<<<grid(DOUT, BS, 1), blk, 0, stream>>>(
      xb, wvT, wv_b, vtmp, BS, DOUT, DIN, 0, 0, 0, 0, 1);

  // latent compressions
  gemm_bf16_wmma<<<grid(DL, BS, 1), blk, 0, stream>>>(
      ktmp, wklT, wkl_b, klat, BS, DL, DOUT, 0, 0, 0, 0, 1);
  gemm_bf16_wmma<<<grid(DL, BS, 1), blk, 0, stream>>>(
      vtmp, wvlT, wvl_b, vlat, BS, DL, DOUT, 0, 0, 0, 0, 1);

  // per-head up-projections (batched over H): K row-major, V transposed
  gemm_bf16_wmma<<<grid(DH, BS, H), blk, 0, stream>>>(
      klat, upkT, upk_b, khead, BS, DH, DL,
      0, (long long)DL * DH, (long long)DH, (long long)BS * DH, 1);
  gemm_bf16_wmma<<<grid(DH, BS, H), blk, 0, stream>>>(
      vlat, upvT, upv_b, vheadT, BS, DH, DL,
      0, (long long)DL * DH, (long long)DH, (long long)BS * DH, 2);

  // causal flash attention
  mla_flash_wmma<<<dim3(S / 64, H, B), dim3(128), 0, stream>>>(
      qb, khead, vheadT, ob, S, BS);

  // output projection (f32 out -> d_out)
  gemm_bf16_wmma<<<grid(DOUT, BS, 1), blk, 0, stream>>>(
      ob, woT, wo_b, d_out, BS, DOUT, DOUT, 0, 0, 0, 0, 0);
}
